// TokenTransformer_86131274154317
// MI455X (gfx1250) — matrix operand order repaired and checked by reading, not compiled, not run on hardware
//
#include <hip/hip_runtime.h>
#include <hip/hip_bf16.h>
#include <math.h>

// ---------------------------------------------------------------------------
// CDNA5 (gfx1250) implementation of the TokenTransformer forward pass.
// All GEMMs (QKV/out projections, MLPs, prediction head) and both attention
// matmuls (QK^T, PV) run on v_wmma_f32_16x16x32_f16 (f16 in, f32 accumulate).
// GEMM is register-blocked 32x64 per wave (8 WMMAs per 32-deep K step from
// 6 fragment loads). Attention is fused flash-style per (batch, head,
// 16-query tile) so the [B,H,N,N] score tensor is never materialized.
// ---------------------------------------------------------------------------

typedef _Float16 h16;
typedef __attribute__((ext_vector_type(16))) _Float16 v16h;
typedef __attribute__((ext_vector_type(8)))  float    v8f;

// ------------------------------ elementwise --------------------------------

__global__ void k_embed(const int* __restrict__ tok, const float* __restrict__ emb,
                        const float* __restrict__ pos, float* __restrict__ x,
                        int M, int D, int Nseq) {
  int i = blockIdx.x * blockDim.x + threadIdx.x;
  if (i >= M * D) return;
  int m = i / D, d = i - m * D;
  int n = m % Nseq;
  x[i] = emb[(size_t)tok[m] * D + d] + pos[(size_t)n * D + d];
}

__global__ void k_tgt_init(const float* __restrict__ qe, const float* __restrict__ pos,
                           float* __restrict__ tgt, h16* __restrict__ tgt16,
                           int M, int D, int Nseq) {
  int i = blockIdx.x * blockDim.x + threadIdx.x;
  if (i >= M * D) return;
  int m = i / D, d = i - m * D;
  int n = m % Nseq;
  float v = qe[(size_t)n * D + d] + pos[(size_t)n * D + d];
  tgt[i] = v;
  tgt16[i] = (h16)v;
}

__global__ void k_f32_to_f16(const float* __restrict__ a, h16* __restrict__ b, int n) {
  int i = blockIdx.x * blockDim.x + threadIdx.x;
  if (i < n) b[i] = (h16)a[i];
}

// W [K,N] f32 row-major  ->  WT [N,K] f16 (so WMMA B-fragments are contiguous)
__global__ void k_wtrans(const float* __restrict__ W, h16* __restrict__ WT,
                         int K, int N) {
  int i = blockIdx.x * blockDim.x + threadIdx.x;
  if (i >= K * N) return;
  int k = i / N, n = i - k * N;
  WT[(size_t)n * K + k] = (h16)W[i];
}

// ------------------------------- layernorm ---------------------------------
// one row per 256-thread block (D == 256); writes f32 and/or f16 outputs
__global__ __launch_bounds__(256)
void k_layernorm(const float* __restrict__ x, const float* __restrict__ g,
                 const float* __restrict__ b, float* __restrict__ y32,
                 h16* __restrict__ y16, int D) {
  __shared__ float red[256];
  int row = blockIdx.x, t = threadIdx.x;
  float v = x[(size_t)row * D + t];
  red[t] = v;
  __syncthreads();
  for (int s = 128; s > 0; s >>= 1) { if (t < s) red[t] += red[t + s]; __syncthreads(); }
  float mean = red[0] / (float)D;
  __syncthreads();
  float c = v - mean;
  red[t] = c * c;
  __syncthreads();
  for (int s = 128; s > 0; s >>= 1) { if (t < s) red[t] += red[t + s]; __syncthreads(); }
  float var = red[0] / (float)D;
  float o = c * rsqrtf(var + 1e-5f) * g[t] + b[t];
  if (y32) y32[(size_t)row * D + t] = o;
  if (y16) y16[(size_t)row * D + t] = (h16)o;
}

// --------------------------------- GEMM ------------------------------------
// C[M,N] = act(A[M,K] @ W[K,N] + bias) (+ resid).  A: f16 rows, WT: f16 [N,K].
// Register-blocked: one wave owns a 32x64 tile = 2 M-tiles x 4 N-tiles.
// Per 32-deep K step: 2 A-fragment + 4 B-fragment loads feed 8 independent
// v_wmma_f32_16x16x32_f16 (no RAW chains -> full XDL pipelining).
// Fragment dword maps used here (wave32; same k order for A and B):
//   A (16x32): lane m=lane&15, half=lane>>4; dwords {0..7}+8*half
//   B (32x16): lane n=lane&15, half=lane>>4; dwords {0..7}+8*half
template<int ACT, int RESID, int OUT16>
__global__ __launch_bounds__(32)
void k_gemm(const h16* __restrict__ A, const h16* __restrict__ WT,
            const float* __restrict__ bias, const float* __restrict__ resid,
            float* __restrict__ out32, h16* __restrict__ out16,
            int Mdim, int Ndim, int Kdim) {
  const int lane = threadIdx.x;
  const int hw = lane >> 4, lm = lane & 15;
  const int tn0 = blockIdx.x * 4;  // first of 4 N-tiles
  const int tm0 = blockIdx.y * 2;  // first of 2 M-tiles
  const uint* ap0 = (const uint*)(A + (size_t)(tm0 * 16 + lm) * Kdim);
  const uint* ap1 = (const uint*)(A + (size_t)(tm0 * 16 + 16 + lm) * Kdim);
  const uint* bp0 = (const uint*)(WT + (size_t)((tn0 + 0) * 16 + lm) * Kdim);
  const uint* bp1 = (const uint*)(WT + (size_t)((tn0 + 1) * 16 + lm) * Kdim);
  const uint* bp2 = (const uint*)(WT + (size_t)((tn0 + 2) * 16 + lm) * Kdim);
  const uint* bp3 = (const uint*)(WT + (size_t)((tn0 + 3) * 16 + lm) * Kdim);
  v8f acc[2][4] = {};
  const int kd = Kdim >> 1;  // dwords per row
  for (int k2 = 0; k2 < kd; k2 += 16) {
    union { uint u[8]; v16h v; } a0, a1, b0, b1, b2, b3;
#pragma unroll
    for (int j = 0; j < 4; ++j) {
      a0.u[j] = ap0[k2 + j + 8 * hw];  a0.u[j + 4] = ap0[k2 + j + 4 + 8 * hw];
      a1.u[j] = ap1[k2 + j + 8 * hw];  a1.u[j + 4] = ap1[k2 + j + 4 + 8 * hw];
      b0.u[j] = bp0[k2 + j + 8 * hw];  b0.u[j + 4] = bp0[k2 + j + 4 + 8 * hw];
      b1.u[j] = bp1[k2 + j + 8 * hw];  b1.u[j + 4] = bp1[k2 + j + 4 + 8 * hw];
      b2.u[j] = bp2[k2 + j + 8 * hw];  b2.u[j + 4] = bp2[k2 + j + 4 + 8 * hw];
      b3.u[j] = bp3[k2 + j + 8 * hw];  b3.u[j + 4] = bp3[k2 + j + 4 + 8 * hw];
    }
    acc[0][0] = __builtin_amdgcn_wmma_f32_16x16x32_f16(false, a0.v, false, b0.v, (short)0, acc[0][0], false, false);
    acc[0][1] = __builtin_amdgcn_wmma_f32_16x16x32_f16(false, a0.v, false, b1.v, (short)0, acc[0][1], false, false);
    acc[0][2] = __builtin_amdgcn_wmma_f32_16x16x32_f16(false, a0.v, false, b2.v, (short)0, acc[0][2], false, false);
    acc[0][3] = __builtin_amdgcn_wmma_f32_16x16x32_f16(false, a0.v, false, b3.v, (short)0, acc[0][3], false, false);
    acc[1][0] = __builtin_amdgcn_wmma_f32_16x16x32_f16(false, a1.v, false, b0.v, (short)0, acc[1][0], false, false);
    acc[1][1] = __builtin_amdgcn_wmma_f32_16x16x32_f16(false, a1.v, false, b1.v, (short)0, acc[1][1], false, false);
    acc[1][2] = __builtin_amdgcn_wmma_f32_16x16x32_f16(false, a1.v, false, b2.v, (short)0, acc[1][2], false, false);
    acc[1][3] = __builtin_amdgcn_wmma_f32_16x16x32_f16(false, a1.v, false, b3.v, (short)0, acc[1][3], false, false);
  }
#pragma unroll
  for (int mt = 0; mt < 2; ++mt) {
#pragma unroll
    for (int nt = 0; nt < 4; ++nt) {
      const int col = (tn0 + nt) * 16 + lm;
      const float bv = bias[col];
#pragma unroll
      for (int r = 0; r < 8; ++r) {
        const int row = (tm0 + mt) * 16 + r + 8 * hw;  // C layout: VGPR r -> M = r + 8*half
        float v = acc[mt][nt][r] + bv;
        if (ACT == 1) {  // GELU (tanh approximation, as jax.nn.gelu default)
          float u = 0.7978845608028654f * (v + 0.044715f * v * v * v);
          v = 0.5f * v * (1.f + tanhf(u));
        } else if (ACT == 2) {
          v = fmaxf(v, 0.f);
        }
        if (RESID) v += resid[(size_t)row * Ndim + col];
        if (OUT16) out16[(size_t)row * Ndim + col] = (h16)v;
        else       out32[(size_t)row * Ndim + col] = v;
      }
    }
  }
}

// ------------------------------ attention ----------------------------------
// Fused flash attention; grid = (Nseq/16 q-tiles, H, B), one wave per block.
// Dh = 32 so QK^T is one WMMA per 16-key group (K-dim = Dh), PV is two WMMAs
// (Dh split in two 16-column tiles) with P staged through LDS in f16 and V
// staged transposed in LDS.
__global__ __launch_bounds__(32)
void k_attention(const h16* __restrict__ qh, const h16* __restrict__ kh,
                 const h16* __restrict__ vh, h16* __restrict__ out,
                 int Nseq, int D, int Dh, int maskX, int maskR, int use_mask) {
  __shared__ h16 P[16 * 32];    // [q-row][key-local]
  __shared__ h16 VT[32 * 34];   // [dh][key-local], padded stride 34
  const int lane = threadIdx.x;
  const int hw = lane >> 4, lm = lane & 15;
  const int qt = blockIdx.x, hd = blockIdx.y, b = blockIdx.z;
  const size_t head_off = (size_t)hd * Dh;

  // Q fragment (A layout), loaded once: row = qt*16 + lm, K-dim = Dh = 32
  union { uint u[8]; v16h v; } qf;
  {
    const uint* qp = (const uint*)(qh + ((size_t)b * Nseq + qt * 16 + lm) * D + head_off);
#pragma unroll
    for (int j = 0; j < 4; ++j) {
      qf.u[j]     = qp[j     + 8 * hw];
      qf.u[j + 4] = qp[j + 4 + 8 * hw];
    }
  }

  float mrun[8], lrun[8];
  v8f o0 = {}, o1 = {};
#pragma unroll
  for (int r = 0; r < 8; ++r) { mrun[r] = -INFINITY; lrun[r] = 0.f; }
  const float scale = rsqrtf((float)Dh);

  for (int kt = 0; kt < Nseq; kt += 32) {
    // stage V tile transposed into LDS: VT[dh][key] ; key = kt + lane
    {
      const h16* vp = vh + ((size_t)b * Nseq + kt + lane) * D + head_off;
#pragma unroll
      for (int d = 0; d < 32; ++d) VT[d * 34 + lane] = vp[d];
    }
    // K fragments (B layout: N = key, K-dim = Dh contiguous per row)
    union { uint u[8]; v16h v; } kf0, kf1;
    {
      const uint* kp0 = (const uint*)(kh + ((size_t)b * Nseq + kt + lm) * D + head_off);
      const uint* kp1 = (const uint*)(kh + ((size_t)b * Nseq + kt + 16 + lm) * D + head_off);
#pragma unroll
      for (int j = 0; j < 8; ++j) { kf0.u[j] = kp0[j + 8 * hw]; kf1.u[j] = kp1[j + 8 * hw]; }
    }
    v8f s0 = __builtin_amdgcn_wmma_f32_16x16x32_f16(false, qf.v, false, kf0.v,
                                                    (short)0, (v8f){}, false, false);
    v8f s1 = __builtin_amdgcn_wmma_f32_16x16x32_f16(false, qf.v, false, kf1.v,
                                                    (short)0, (v8f){}, false, false);
    // online softmax; row m lives on one 16-lane half => shfl_xor 8/4/2/1
#pragma unroll
    for (int r = 0; r < 8; ++r) {
      const int qrow = qt * 16 + r + 8 * hw;
      float a = s0[r] * scale, c = s1[r] * scale;
      if (use_mask) {
        const int xq = qrow / maskX, zq = qrow % maskX;
        const int k0 = kt + lm, k1 = kt + 16 + lm;
        const int x0 = k0 / maskX, z0 = k0 % maskX;
        const int x1 = k1 / maskX, z1 = k1 % maskX;
        if (abs(xq - x0) > maskR || abs(zq - z0) > maskR) a += -1e9f;
        if (abs(xq - x1) > maskR || abs(zq - z1) > maskR) c += -1e9f;
      }
      float mx = fmaxf(a, c);
      for (int d = 8; d > 0; d >>= 1) mx = fmaxf(mx, __shfl_xor(mx, d, 32));
      const float mnew = fmaxf(mrun[r], mx);
      const float corr = __expf(mrun[r] - mnew);
      const float p0 = __expf(a - mnew), p1 = __expf(c - mnew);
      float sum = p0 + p1;
      for (int d = 8; d > 0; d >>= 1) sum += __shfl_xor(sum, d, 32);
      lrun[r] = lrun[r] * corr + sum;
      mrun[r] = mnew;
      o0[r] *= corr; o1[r] *= corr;
      const int prow = r + 8 * hw;
      P[prow * 32 + lm]      = (h16)p0;
      P[prow * 32 + 16 + lm] = (h16)p1;
    }
    __syncthreads();
    // P fragment (A layout, K = 32 keys) and V fragments (B layout from VT)
    union { uint u[8]; v16h v; } pf, vf0, vf1;
    {
      const uint* pp = (const uint*)(P + lm * 32);
#pragma unroll
      for (int j = 0; j < 4; ++j) {
        pf.u[j]     = pp[j     + 8 * hw];
        pf.u[j + 4] = pp[j + 4 + 8 * hw];
      }
      const uint* vt0 = (const uint*)(VT + lm * 34);
      const uint* vt1 = (const uint*)(VT + (16 + lm) * 34);
#pragma unroll
      for (int j = 0; j < 8; ++j) { vf0.u[j] = vt0[j + 8 * hw]; vf1.u[j] = vt1[j + 8 * hw]; }
    }
    o0 = __builtin_amdgcn_wmma_f32_16x16x32_f16(false, pf.v, false, vf0.v,
                                                (short)0, o0, false, false);
    o1 = __builtin_amdgcn_wmma_f32_16x16x32_f16(false, pf.v, false, vf1.v,
                                                (short)0, o1, false, false);
    __syncthreads();
  }
#pragma unroll
  for (int r = 0; r < 8; ++r) {
    const int qrow = qt * 16 + r + 8 * hw;
    const float inv = 1.f / lrun[r];
    h16* op = out + ((size_t)b * Nseq + qrow) * D + head_off;
    op[lm]      = (h16)(o0[r] * inv);
    op[16 + lm] = (h16)(o1[r] * inv);
  }
}

// ------------------------------ orchestration ------------------------------

extern "C" void kernel_launch(void* const* d_in, const int* in_sizes, int n_in,
                              void* d_out, int out_size, void* d_ws, size_t ws_size,
                              hipStream_t stream) {
  (void)out_size; (void)ws_size;
  const int D = 256, F = 1024, H = 8, Dh = 32;
  const int Nd = in_sizes[2] / D;     // 1600
  const int M  = in_sizes[0];        // B*N = 6400
  const int Bb = M / Nd;             // 4
  const int CB = in_sizes[n_in - 1]; // 256 (pred_b)
  const int X  = (int)(sqrtf((float)Nd) + 0.5f);  // 40
  const int R  = 7 / 2;                           // kernel radius 3

  // --- workspace carve-up ---
  char* p = (char*)d_ws;
  auto alloc = [&](size_t bytes) -> void* {
    void* r = (void*)p; p += (bytes + 255) & ~(size_t)255; return r;
  };
  float* x    = (float*)alloc((size_t)M * D * 4);
  float* tmp  = (float*)alloc((size_t)M * D * 4);
  float* tgt  = (float*)alloc((size_t)M * D * 4);
  h16*   hA   = (h16*)alloc((size_t)M * F * 2);
  h16*   hB   = (h16*)alloc((size_t)M * F * 2);
  h16*   qh   = (h16*)alloc((size_t)M * D * 2);
  h16*   kh   = (h16*)alloc((size_t)M * D * 2);
  h16*   vh   = (h16*)alloc((size_t)M * D * 2);
  h16*   aoh  = (h16*)alloc((size_t)M * D * 2);
  h16*   memh = (h16*)alloc((size_t)M * D * 2);
  h16*   w16  = (h16*)alloc((size_t)D * F * 2);

  auto FP = [&](const void* q) { return (const float*)q; };
  auto wtrans = [&](const void* W, int K, int N) {
    int n = K * N;
    k_wtrans<<<(n + 255) / 256, 256, 0, stream>>>(FP(W), w16, K, N);
  };
  auto gemm = [&](const h16* A, const void* bias, int Md, int Ndim, int Kd,
                  int act, const float* resid, float* o32, h16* o16) {
    dim3 g(Ndim / 64, Md / 32);
    const float* bp = FP(bias);
    if (o16) {
      if (act == 1)      k_gemm<1,0,1><<<g,32,0,stream>>>(A, w16, bp, nullptr, nullptr, o16, Md, Ndim, Kd);
      else if (act == 2) k_gemm<2,0,1><<<g,32,0,stream>>>(A, w16, bp, nullptr, nullptr, o16, Md, Ndim, Kd);
      else               k_gemm<0,0,1><<<g,32,0,stream>>>(A, w16, bp, nullptr, nullptr, o16, Md, Ndim, Kd);
    } else if (resid) {
      k_gemm<0,1,0><<<g,32,0,stream>>>(A, w16, bp, resid, o32, nullptr, Md, Ndim, Kd);
    } else {
      k_gemm<0,0,0><<<g,32,0,stream>>>(A, w16, bp, nullptr, o32, nullptr, Md, Ndim, Kd);
    }
  };
  auto ln = [&](const float* in, const void* g_, const void* b_, float* o32, h16* o16) {
    k_layernorm<<<M, 256, 0, stream>>>(in, FP(g_), FP(b_), o32, o16, D);
  };
  auto attn = [&](int use_mask) {
    dim3 g(Nd / 16, H, Bb);
    k_attention<<<g, 32, 0, stream>>>(qh, kh, vh, aoh, Nd, D, Dh, X, R, use_mask);
  };

  // --- embedding ---
  k_embed<<<(M * D + 255) / 256, 256, 0, stream>>>(
      (const int*)d_in[0], FP(d_in[1]), FP(d_in[2]), x, M, D, Nd);

  // --- encoder: 2 pre-norm ViT blocks ---
  for (int l = 0; l < 2; ++l) {
    void* const* L = d_in + 4 + 16 * l;
    // ln1_g ln1_b Wq bq Wk bk Wv bv Wo bo ln2_g ln2_b W1 b1 W2 b2
    ln(x, L[0], L[1], nullptr, hA);
    wtrans(L[2], D, D);  gemm(hA, L[3], M, D, D, 0, nullptr, nullptr, qh);
    wtrans(L[4], D, D);  gemm(hA, L[5], M, D, D, 0, nullptr, nullptr, kh);
    wtrans(L[6], D, D);  gemm(hA, L[7], M, D, D, 0, nullptr, nullptr, vh);
    attn(0);
    wtrans(L[8], D, D);  gemm(aoh, L[9], M, D, D, 0, x, x, nullptr);      // x += attn proj
    ln(x, L[10], L[11], nullptr, hA);
    wtrans(L[12], D, F); gemm(hA, L[13], M, F, D, 1, nullptr, nullptr, hB); // gelu
    wtrans(L[14], F, D); gemm(hB, L[15], M, D, F, 0, x, x, nullptr);      // x += mlp
  }
  // mem = x
  k_f32_to_f16<<<(M * D + 255) / 256, 256, 0, stream>>>(x, memh, M * D);
  // tgt = query_emb + pos_emb
  k_tgt_init<<<(M * D + 255) / 256, 256, 0, stream>>>(FP(d_in[3]), FP(d_in[2]),
                                                      tgt, hA, M, D, Nd);

  // --- decoder: 4 post-norm DETR blocks ---
  for (int l = 0; l < 4; ++l) {
    void* const* L = d_in + 4 + 32 + 26 * l;
    // sWq sbq sWk sbk sWv sbv sWo sbo ln1_g ln1_b cWq cbq cWk cbk cWv cbv cWo cbo
    // ln2_g ln2_b W1 b1 W2 b2 ln3_g ln3_b
    // hA holds f16(tgt)
    wtrans(L[0], D, D);  gemm(hA, L[1], M, D, D, 0, nullptr, nullptr, qh);
    wtrans(L[2], D, D);  gemm(hA, L[3], M, D, D, 0, nullptr, nullptr, kh);
    wtrans(L[4], D, D);  gemm(hA, L[5], M, D, D, 0, nullptr, nullptr, vh);
    attn(1);                                                    // banded mask
    wtrans(L[6], D, D);  gemm(aoh, L[7], M, D, D, 0, tgt, tmp, nullptr);
    ln(tmp, L[8], L[9], tgt, hA);                               // tgt = LN(tgt + t2)
    wtrans(L[10], D, D); gemm(hA, L[11], M, D, D, 0, nullptr, nullptr, qh);
    wtrans(L[12], D, D); gemm(memh, L[13], M, D, D, 0, nullptr, nullptr, kh);
    wtrans(L[14], D, D); gemm(memh, L[15], M, D, D, 0, nullptr, nullptr, vh);
    attn(0);                                                    // cross-attn
    wtrans(L[16], D, D); gemm(aoh, L[17], M, D, D, 0, tgt, tmp, nullptr);
    ln(tmp, L[18], L[19], tgt, hA);
    wtrans(L[20], D, F); gemm(hA, L[21], M, F, D, 2, nullptr, nullptr, hB); // relu
    wtrans(L[22], F, D); gemm(hB, L[23], M, D, F, 0, tgt, tmp, nullptr);
    ln(tmp, L[24], L[25], tgt, hA);
  }

  // --- head: LN(tgt) @ pred_W + pred_b ---
  const int base = 4 + 32 + 26 * 4;  // 140
  ln(tgt, d_in[base], d_in[base + 1], nullptr, hA);
  wtrans(d_in[base + 2], D, CB);
  gemm(hA, d_in[base + 3], M, CB, D, 0, nullptr, (float*)d_out, nullptr);
}
